// GatedDeltaNet_84868553769177
// MI455X (gfx1250) — matrix operand order repaired and checked by reading, not compiled, not run on hardware
//
#include <hip/hip_runtime.h>
#include <hip/hip_bf16.h>
#include <math.h>
#include <stdint.h>

#define B_  2
#define T_  1024
#define HID 1024
#define NH  6
#define DK  128
#define DV  256
#define KD  768     // NH*DK
#define VD  1536    // NH*DV
#define M_  (B_*T_) // 2048
#define CONVK 4

typedef __attribute__((ext_vector_type(16))) __bf16 v16bf;
typedef __attribute__((ext_vector_type(8)))  float  v8f;
typedef __attribute__((ext_vector_type(4)))  unsigned int u32x4;
typedef __attribute__((ext_vector_type(8)))  int i32x8;
typedef __attribute__((ext_vector_type(4)))  int i32x4;

#if __has_builtin(__builtin_amdgcn_tensor_load_to_lds)
#define HAVE_TDM 1
#endif
#if __has_include(<hip/amd_detail/amd_gfx1250_TDM.h>)
#define TDM_ARGS6 1
#endif

__device__ __forceinline__ unsigned short f2bf(float f) {
  union { float f; unsigned u; } x; x.f = f;
  unsigned u = x.u;
  u += 0x7FFFu + ((u >> 16) & 1u);   // round-to-nearest-even
  return (unsigned short)(u >> 16);
}

union FragBF { v16bf v; unsigned short s[16]; };

#if HAVE_TDM
// ---------------------------------------------------------------------------
// TDM: async DMA of a (tile_rows x 32) bf16 tile (row-major, row stride = K
// elements) from global into LDS (packed, stride 32). D# per CDNA5 ISA ch.8.
// Issued by one wave; tracked with TENSORcnt.
// ---------------------------------------------------------------------------
__device__ __forceinline__ void tdm_load_2d(unsigned lds_byte,
                                            const unsigned short* gptr,
                                            int K, int rows_total,
                                            int tile_rows) {
  unsigned long long ga = (unsigned long long)(uintptr_t)gptr;
  u32x4 g0;
  g0[0] = 1u;                                   // count=1 (valid descriptor)
  g0[1] = lds_byte;                             // lds_addr [63:32]
  g0[2] = (unsigned)(ga & 0xFFFFFFFFu);         // global_addr lo
  g0[3] = (unsigned)((ga >> 32) & 0x01FFFFFFu)  // global_addr hi (57-bit)
          | 0x80000000u;                        // type=2 ("image") [127:126]
  i32x8 g1;
  g1[0] = (int)(1u << 16);                      // data_size=1 -> 2 bytes
  g1[1] = (int)(((unsigned)K & 0xFFFFu) << 16); // tensor_dim0[15:0] @ [63:48]
  g1[2] = (int)((((unsigned)K) >> 16) |
                (((unsigned)rows_total & 0xFFFFu) << 16)); // dim0 hi | dim1 lo
  g1[3] = (int)((((unsigned)rows_total) >> 16) |
                (32u << 16));                   // tensor_dim1 hi | tile_dim0=32
  g1[4] = tile_rows;                            // tile_dim1 ; tile_dim2=0
  g1[5] = K;                                    // tensor_dim0_stride[31:0]
  g1[6] = 0;                                    // stride hi | dim1_stride lo
  g1[7] = 0;
  i32x4 z4 = {0, 0, 0, 0};
#if TDM_ARGS6
  i32x8 z8 = {0, 0, 0, 0, 0, 0, 0, 0};
  __builtin_amdgcn_tensor_load_to_lds(g0, g1, z4, z4, z8, 0);
#else
  __builtin_amdgcn_tensor_load_to_lds(g0, g1, z4, z4, 0);
#endif
}
#endif

// ---------------------------------------------------------------------------
// C[M,N] = A[M,K] @ W[N,K]^T ; A,W pre-converted bf16, fp32 accum via WMMA.
// Block: 256 threads (8 wave32). Block tile 128x64, K-step 32.
// Double-buffered LDS tiles fed by the Tensor Data Mover (wave 0 issues,
// waits on TENSORcnt, workgroup barrier publishes).
// ---------------------------------------------------------------------------
__global__ __launch_bounds__(256) void k_gemm_tdm(
    const unsigned short* __restrict__ A, const unsigned short* __restrict__ W,
    float* __restrict__ C, int M, int N, int K) {
  __shared__ unsigned short As[2][128 * 32];
  __shared__ unsigned short Ws[2][64 * 32];

  const int tid  = threadIdx.x;
  const int lane = tid & 31;
  const int wv   = tid >> 5;
  const int hi   = lane >> 4;
  const int lo   = lane & 15;
  const int row0 = blockIdx.y * 128;
  const int col0 = blockIdx.x * 64;
  const int nk   = K >> 5;

  v8f acc[4];
  #pragma unroll
  for (int nt = 0; nt < 4; ++nt)
    #pragma unroll
    for (int e = 0; e < 8; ++e) acc[nt][e] = 0.f;

#if HAVE_TDM
  if (tid < 32) {   // wave 0: prologue DMA of tile 0
    tdm_load_2d((unsigned)(uintptr_t)&As[0][0], A + (size_t)row0 * K, K, M, 128);
    tdm_load_2d((unsigned)(uintptr_t)&Ws[0][0], W + (size_t)col0 * K, K, N, 64);
  }
#endif

  for (int kb = 0; kb < nk; ++kb) {
    const int cur = kb & 1;
#if HAVE_TDM
    if (tid < 32) {
      if (kb + 1 < nk) {  // issue next tile into the other buffer, then wait
        tdm_load_2d((unsigned)(uintptr_t)&As[cur ^ 1][0],
                    A + (size_t)row0 * K + (size_t)(kb + 1) * 32, K, M, 128);
        tdm_load_2d((unsigned)(uintptr_t)&Ws[cur ^ 1][0],
                    W + (size_t)col0 * K + (size_t)(kb + 1) * 32, K, N, 64);
        __builtin_amdgcn_s_wait_tensorcnt(2);   // current tile's 2 ops done
      } else {
        __builtin_amdgcn_s_wait_tensorcnt(0);
      }
    }
    __syncthreads();
#else
    // fallback: cooperative vector copy (16B chunks)
    for (int i = tid; i < 128 * 4; i += 256) {
      int r = i >> 2, c = (i & 3) * 8;
      *(uint4*)&As[cur][r * 32 + c] =
          *(const uint4*)&A[(size_t)(row0 + r) * K + kb * 32 + c];
    }
    for (int i = tid; i < 64 * 4; i += 256) {
      int r = i >> 2, c = (i & 3) * 8;
      *(uint4*)&Ws[cur][r * 32 + c] =
          *(const uint4*)&W[(size_t)(col0 + r) * K + kb * 32 + c];
    }
    __syncthreads();
#endif

    // A fragment: 16x32 bf16, K = hi*16 + i (lanes 0-15/16-31 both M=lo)
    FragBF a;
    const int mrow = wv * 16 + lo;
    #pragma unroll
    for (int i = 0; i < 16; ++i) {
      int kk = hi * 16 + i;
      a.s[i] = As[cur][mrow * 32 + kk];
    }

    #pragma unroll
    for (int nt = 0; nt < 4; ++nt) {
      FragBF bfr;  // B fragment: lane -> N=lo, K = hi*16 + i
      const int ncol = nt * 16 + lo;
      #pragma unroll
      for (int i = 0; i < 16; ++i) bfr.s[i] = Ws[cur][ncol * 32 + hi * 16 + i];
      acc[nt] = __builtin_amdgcn_wmma_f32_16x16x32_bf16(
          false, a.v, false, bfr.v, (short)0, acc[nt], false, false);
    }
    __syncthreads();
  }

  #pragma unroll
  for (int nt = 0; nt < 4; ++nt) {
    #pragma unroll
    for (int r = 0; r < 8; ++r) {
      int mg = row0 + wv * 16 + r + 8 * hi;
      int ng = col0 + nt * 16 + lo;
      C[(size_t)mg * N + ng] = acc[nt][r];
    }
  }
}

// ---------------------------------------------------------------------------
// fp32 -> bf16 (RNE), 4 elements/thread, packed 64-bit stores
// ---------------------------------------------------------------------------
__global__ __launch_bounds__(256) void k_cvt_bf16(
    const float* __restrict__ s, unsigned short* __restrict__ d, int n) {
  const int i = (blockIdx.x * 256 + threadIdx.x) * 4;
  if (i >= n) return;
  float4 f = *(const float4*)(s + i);
  uint2 p;
  p.x = (unsigned)f2bf(f.x) | ((unsigned)f2bf(f.y) << 16);
  p.y = (unsigned)f2bf(f.z) | ((unsigned)f2bf(f.w) << 16);
  *(uint2*)(d + i) = p;
}

// ---------------------------------------------------------------------------
// beta = sigmoid(hs @ b_w.T); g = -exp(A_log)*softplus(hs @ a_w.T + dt_bias)
// ---------------------------------------------------------------------------
__global__ __launch_bounds__(256) void k_small_proj(
    const float* __restrict__ hs, const float* __restrict__ a_w,
    const float* __restrict__ b_w, const float* __restrict__ A_log,
    const float* __restrict__ dt_bias,
    float* __restrict__ g_buf, float* __restrict__ beta_buf) {
  __shared__ float red[12 * 256];
  const int bt = blockIdx.x, tid = threadIdx.x;
  const float* row = hs + (size_t)bt * HID;
  float pa[NH], pb[NH];
  #pragma unroll
  for (int h = 0; h < NH; ++h) { pa[h] = 0.f; pb[h] = 0.f; }
  for (int i = tid; i < HID; i += 256) {
    float x = row[i];
    #pragma unroll
    for (int h = 0; h < NH; ++h) {
      pa[h] += x * a_w[h * HID + i];
      pb[h] += x * b_w[h * HID + i];
    }
  }
  #pragma unroll
  for (int h = 0; h < NH; ++h) {
    red[h * 256 + tid]        = pa[h];
    red[(NH + h) * 256 + tid] = pb[h];
  }
  __syncthreads();
  for (int s = 128; s > 0; s >>= 1) {
    if (tid < s)
      for (int j = 0; j < 12; ++j) red[j * 256 + tid] += red[j * 256 + tid + s];
    __syncthreads();
  }
  if (tid < NH) {
    float da = red[tid * 256];
    float db = red[(NH + tid) * 256];
    float x  = da + dt_bias[tid];
    float sp = (x > 20.f) ? x : log1pf(__expf(x));
    g_buf[bt * NH + tid]    = -__expf(A_log[tid]) * sp;
    beta_buf[bt * NH + tid] = 1.f / (1.f + __expf(-db));
  }
}

// ---------------------------------------------------------------------------
// depthwise causal conv1d (K=4) + SiLU ; x,y: [B,T,D], w: [D,4]
// ---------------------------------------------------------------------------
__global__ __launch_bounds__(256) void k_conv_silu(
    const float* __restrict__ x, const float* __restrict__ w,
    float* __restrict__ y, int D) {
  const int flat = blockIdx.x * 256 + threadIdx.x;
  if (flat >= M_ * D) return;
  const int d  = flat % D;
  const int bt = flat / D;
  const int t  = bt % T_;
  float acc = 0.f;
  #pragma unroll
  for (int j = 0; j < CONVK; ++j) {
    int tt = t - (CONVK - 1) + j;
    if (tt >= 0) acc += x[(size_t)(bt - (t - tt)) * D + d] * w[d * CONVK + j];
  }
  y[(size_t)bt * D + d] = acc / (1.f + __expf(-acc));
}

// ---------------------------------------------------------------------------
// per-(b,t,h): q = l2norm(q)*dk^-0.5 ; k = l2norm(k) ; in place
// ---------------------------------------------------------------------------
__global__ __launch_bounds__(128) void k_l2norm(
    float* __restrict__ q, float* __restrict__ k) {
  __shared__ float rq[128], rk[128];
  const int h  = blockIdx.x % NH;
  const int bt = blockIdx.x / NH;
  const int tid = threadIdx.x;
  float* qp = q + (size_t)bt * KD + h * DK;
  float* kp = k + (size_t)bt * KD + h * DK;
  float qv = qp[tid], kv = kp[tid];
  rq[tid] = qv * qv; rk[tid] = kv * kv;
  __syncthreads();
  for (int s = 64; s > 0; s >>= 1) {
    if (tid < s) { rq[tid] += rq[tid + s]; rk[tid] += rk[tid + s]; }
    __syncthreads();
  }
  const float qs = rsqrtf(rq[0] + 1e-6f) * 0.08838834764831845f; // 128^-0.5
  const float ks = rsqrtf(rk[0] + 1e-6f);
  qp[tid] = qv * qs;
  kp[tid] = kv * ks;
}

// ---------------------------------------------------------------------------
// gated delta rule scan. One block per (b,h); 256 threads = 256 v-columns.
// Each thread keeps S[:,vv] (128 fp32) in VGPRs (fully unrolled).
// ---------------------------------------------------------------------------
__global__ __launch_bounds__(256) void k_delta_rule(
    const float* __restrict__ q, const float* __restrict__ k,
    const float* __restrict__ v, const float* __restrict__ g_buf,
    const float* __restrict__ beta_buf, float* __restrict__ o_buf) {
  __shared__ float qs[DK], ks[DK];
  const int b = blockIdx.x / NH;
  const int h = blockIdx.x % NH;
  const int tid = threadIdx.x;

  float S[DK];
  #pragma unroll
  for (int kk = 0; kk < DK; ++kk) S[kk] = 0.f;

  for (int t = 0; t < T_; ++t) {
    const int bt = b * T_ + t;
    if (tid < DK) {
      qs[tid] = q[(size_t)bt * KD + h * DK + tid];
      ks[tid] = k[(size_t)bt * KD + h * DK + tid];
    }
    __syncthreads();

    const float gt = g_buf[bt * NH + h];
    const float bb = beta_buf[bt * NH + h];
    const float eg = __expf(gt);
    const float vv = v[(size_t)bt * VD + h * DV + tid];

    float kv = 0.f;
    #pragma unroll
    for (int kk = 0; kk < DK; ++kk) kv += ks[kk] * S[kk];
    kv *= eg;
    const float delta = (vv - kv) * bb;

    float o = 0.f;
    #pragma unroll
    for (int kk = 0; kk < DK; ++kk) {
      S[kk] = S[kk] * eg + ks[kk] * delta;
      o += qs[kk] * S[kk];
    }
    o_buf[(size_t)bt * VD + h * DV + tid] = o;
    __syncthreads();
  }
}

// ---------------------------------------------------------------------------
// o = o * silu(gate); RMSNorm(eps=1e-5) * norm_weight ; writes bf16 output
// ---------------------------------------------------------------------------
__global__ __launch_bounds__(256) void k_gate_norm(
    const float* __restrict__ o_buf, const float* __restrict__ gate,
    const float* __restrict__ nw, unsigned short* __restrict__ o_bf) {
  __shared__ float red[256];
  const int bt = blockIdx.x, tid = threadIdx.x;
  float og[VD / 256];
  float ss = 0.f;
  #pragma unroll
  for (int j = 0; j < VD / 256; ++j) {
    int d = tid + j * 256;
    float o  = o_buf[(size_t)bt * VD + d];
    float gv = gate[(size_t)bt * VD + d];
    float val = o * (gv / (1.f + __expf(-gv)));
    og[j] = val;
    ss += val * val;
  }
  red[tid] = ss;
  __syncthreads();
  for (int s = 128; s > 0; s >>= 1) {
    if (tid < s) red[tid] += red[tid + s];
    __syncthreads();
  }
  const float r = rsqrtf(red[0] / (float)VD + 1e-5f);
  #pragma unroll
  for (int j = 0; j < VD / 256; ++j) {
    int d = tid + j * 256;
    o_bf[(size_t)bt * VD + d] = f2bf(og[j] * r * nw[d]);
  }
}

// ---------------------------------------------------------------------------
extern "C" void kernel_launch(void* const* d_in, const int* in_sizes, int n_in,
                              void* d_out, int out_size, void* d_ws, size_t ws_size,
                              hipStream_t stream) {
  const float* hs      = (const float*)d_in[0];
  const float* q_w     = (const float*)d_in[1];
  const float* k_w     = (const float*)d_in[2];
  const float* v_w     = (const float*)d_in[3];
  const float* a_w     = (const float*)d_in[4];
  const float* b_w     = (const float*)d_in[5];
  const float* g_w     = (const float*)d_in[6];
  const float* o_w     = (const float*)d_in[7];
  const float* cq_w    = (const float*)d_in[8];
  const float* ck_w    = (const float*)d_in[9];
  const float* cv_w    = (const float*)d_in[10];
  const float* A_log   = (const float*)d_in[11];
  const float* dt_bias = (const float*)d_in[12];
  const float* norm_w  = (const float*)d_in[13];
  float* out = (float*)d_out;

  // fp32 workspace
  float* ws = (float*)d_ws;
  size_t off = 0;
  float* q_pre = ws + off; off += (size_t)M_ * KD;
  float* k_pre = ws + off; off += (size_t)M_ * KD;
  float* v_pre = ws + off; off += (size_t)M_ * VD;
  float* gatep = ws + off; off += (size_t)M_ * VD;
  float* qb    = ws + off; off += (size_t)M_ * KD;
  float* kb    = ws + off; off += (size_t)M_ * KD;
  float* vb    = ws + off; off += (size_t)M_ * VD;
  float* g_buf = ws + off; off += (size_t)M_ * NH;
  float* beta  = ws + off; off += (size_t)M_ * NH;
  float* o_buf = ws + off; off += (size_t)M_ * VD;
  // bf16 (ushort) workspace
  unsigned short* us = (unsigned short*)(ws + off);
  size_t uoff = 0;
  unsigned short* hs_bf = us + uoff; uoff += (size_t)M_ * HID;
  unsigned short* qw_bf = us + uoff; uoff += (size_t)KD * HID;
  unsigned short* kw_bf = us + uoff; uoff += (size_t)KD * HID;
  unsigned short* vw_bf = us + uoff; uoff += (size_t)VD * HID;
  unsigned short* gw_bf = us + uoff; uoff += (size_t)VD * HID;
  unsigned short* ow_bf = us + uoff; uoff += (size_t)HID * VD;
  unsigned short* o_bf  = us + uoff; uoff += (size_t)M_ * VD;

  dim3 blk(256);

  // bf16 operand preparation
  k_cvt_bf16<<<dim3((M_ * HID) / 1024), blk, 0, stream>>>(hs,  hs_bf, M_ * HID);
  k_cvt_bf16<<<dim3((KD * HID) / 1024), blk, 0, stream>>>(q_w, qw_bf, KD * HID);
  k_cvt_bf16<<<dim3((KD * HID) / 1024), blk, 0, stream>>>(k_w, kw_bf, KD * HID);
  k_cvt_bf16<<<dim3((VD * HID) / 1024), blk, 0, stream>>>(v_w, vw_bf, VD * HID);
  k_cvt_bf16<<<dim3((VD * HID) / 1024), blk, 0, stream>>>(g_w, gw_bf, VD * HID);
  k_cvt_bf16<<<dim3((HID * VD) / 1024), blk, 0, stream>>>(o_w, ow_bf, HID * VD);

  // a/b head projections -> g, beta
  k_small_proj<<<dim3(M_), blk, 0, stream>>>(hs, a_w, b_w, A_log, dt_bias, g_buf, beta);

  // big input projections (TDM-staged bf16 WMMA)
  k_gemm_tdm<<<dim3(KD / 64, M_ / 128), blk, 0, stream>>>(hs_bf, qw_bf, q_pre, M_, KD, HID);
  k_gemm_tdm<<<dim3(KD / 64, M_ / 128), blk, 0, stream>>>(hs_bf, kw_bf, k_pre, M_, KD, HID);
  k_gemm_tdm<<<dim3(VD / 64, M_ / 128), blk, 0, stream>>>(hs_bf, vw_bf, v_pre, M_, VD, HID);
  k_gemm_tdm<<<dim3(VD / 64, M_ / 128), blk, 0, stream>>>(hs_bf, gw_bf, gatep, M_, VD, HID);

  // causal depthwise conv + SiLU
  k_conv_silu<<<dim3((M_ * KD + 255) / 256), blk, 0, stream>>>(q_pre, cq_w, qb, KD);
  k_conv_silu<<<dim3((M_ * KD + 255) / 256), blk, 0, stream>>>(k_pre, ck_w, kb, KD);
  k_conv_silu<<<dim3((M_ * VD + 255) / 256), blk, 0, stream>>>(v_pre, cv_w, vb, VD);

  // q/k l2 normalization (+ q scale)
  k_l2norm<<<dim3(M_ * NH), dim3(128), 0, stream>>>(qb, kb);

  // sequential gated delta rule
  k_delta_rule<<<dim3(B_ * NH), blk, 0, stream>>>(qb, kb, vb, g_buf, beta, o_buf);

  // gate + RMSNorm -> bf16
  k_gate_norm<<<dim3(M_), blk, 0, stream>>>(o_buf, gatep, norm_w, o_bf);

  // output projection (TDM-staged bf16 WMMA)
  k_gemm_tdm<<<dim3(HID / 64, M_ / 128), blk, 0, stream>>>(o_bf, ow_bf, out, M_, HID, VD);
}
